// MeanPool_73194832658764
// MI455X (gfx1250) — compile-verified, matrix operand order checked by reading, not run on hardware
//
#include <hip/hip_runtime.h>
#include <hip/hip_bf16.h>

typedef __attribute__((ext_vector_type(16))) __bf16 v16bf;
typedef __attribute__((ext_vector_type(8)))  float  v8f;
typedef __attribute__((ext_vector_type(4)))  float  f32x4;

#define D_NODE 256
#define D_EDGE 64
#define HDIM   128
#define D_MSG  256
#define D_OUT  256

union FragB16 {
    uint4 u[2];
    v16bf v;
};

// ---- f32 -> bf16 conversion via native fptrunc (RNE) ----
__device__ __forceinline__ unsigned short bf16s(float f) {
    union { __bf16 h; unsigned short u; } c;
    c.h = (__bf16)f;
    return c.u;
}
__device__ __forceinline__ unsigned int bf16x2(float a, float b) {
    union { __bf16 h[2]; unsigned int u; } c;
    c.h[0] = (__bf16)a;
    c.h[1] = (__bf16)b;
    return c.u;
}
// non-temporal vector load + convert (for single-use streams: x, ez)
__device__ __forceinline__ uint2 cvt4_nt(const float* p) {
    f32x4 f = __builtin_nontemporal_load((const f32x4*)p);
    return make_uint2(bf16x2(f.x, f.y), bf16x2(f.z, f.w));
}
__device__ __forceinline__ uint4 cvt8_nt(const float* p) {
    f32x4 f0 = __builtin_nontemporal_load((const f32x4*)(p));
    f32x4 f1 = __builtin_nontemporal_load((const f32x4*)(p + 4));
    return make_uint4(bf16x2(f0.x, f0.y), bf16x2(f0.z, f0.w),
                      bf16x2(f1.x, f1.y), bf16x2(f1.z, f1.w));
}

__global__ void zero_f32(float* p, long n) {
    long i = (long)blockIdx.x * blockDim.x + threadIdx.x;
    long s = (long)gridDim.x * blockDim.x;
    for (; i < n; i += s) p[i] = 0.0f;
}

__global__ void cvt_bf16(const float* __restrict__ src, unsigned short* __restrict__ dst, int n) {
    int i = blockIdx.x * blockDim.x + threadIdx.x;
    int s = gridDim.x * blockDim.x;
    for (; i < n; i += s) dst[i] = bf16s(src[i]);
}

// hn = x @ Wn.T + bn   [N,128].  Block: 16 rows staged in LDS; 8 waves = 8 col-tiles.
__global__ __launch_bounds__(256) void hn_gemm(
    const float* __restrict__ x, const unsigned short* __restrict__ WnB,
    const float* __restrict__ bn, float* __restrict__ hn, int nNodes) {
    __shared__ unsigned short SA[16 * D_NODE];          // 16 x 256 bf16 = 8KB
    const int tid  = threadIdx.x;
    const int wave = tid >> 5;
    const int lane = tid & 31;
    const int rl   = lane & 15;
    const int half = lane >> 4;
    const int rowBase = blockIdx.x * 16;

    // stage + convert A tile once per block (thread t: row t/16, 16 cols), NT stream
    {
        const int r  = tid >> 4;
        const int c0 = (tid & 15) * 16;
        const int node = (rowBase + r < nNodes) ? rowBase + r : nNodes - 1;
        const float* p = x + (long)node * D_NODE + c0;
        *(uint4*)(SA + r * D_NODE + c0)     = cvt8_nt(p);
        *(uint4*)(SA + r * D_NODE + c0 + 8) = cvt8_nt(p + 8);
    }
    __syncthreads();

    const int n = wave * 16 + rl;                       // output column 0..127
    const unsigned short* wr = WnB + n * D_NODE;
    const unsigned short* sr = SA + rl * D_NODE;

    v8f acc = {};
#pragma unroll
    for (int kk = 0; kk < D_NODE; kk += 32) {
        FragB16 a, b;
        a.u[0] = *(const uint4*)(sr + kk + 8 * half);
        a.u[1] = *(const uint4*)(sr + kk + 16 + 8 * half);
        b.u[0] = *(const uint4*)(wr + kk + 8 * half);
        b.u[1] = *(const uint4*)(wr + kk + 16 + 8 * half);
        acc = __builtin_amdgcn_wmma_f32_16x16x32_bf16(false, a.v, false, b.v,
                                                      (short)0, acc, false, false);
    }
    const float bias = bn[n];
#pragma unroll
    for (int r = 0; r < 8; ++r) {
        int orow = rowBase + r + 8 * half;
        if (orow < nNodes) hn[(long)orow * HDIM + n] = acc[r] + bias;   // RT: reused
    }
}

// Fused: he = ez @ We.T + be for 16 edges (A tile staged in LDS), scattered to
// sum_e[dst]; plus gather hn[src] -> atomic add into sum_n[dst]; plus deg += 1.
__global__ __launch_bounds__(256) void edge_msg(
    const float* __restrict__ ez, const unsigned short* __restrict__ WeB,
    const float* __restrict__ be, const int* __restrict__ src, const int* __restrict__ dst,
    const float* __restrict__ hn, float* __restrict__ sum_n, float* __restrict__ sum_e,
    float* __restrict__ deg, long nEdges) {
    __shared__ unsigned short SE[16 * D_EDGE];          // 16 x 64 bf16 = 2KB
    const int tid  = threadIdx.x;
    const int wave = tid >> 5;
    const int lane = tid & 31;
    const int rl   = lane & 15;
    const int half = lane >> 4;
    const long edgeBase = (long)blockIdx.x * 16;

    // stage + convert ez tile once per block (thread t: row t/16, 4 cols), NT stream
    {
        const int r  = tid >> 4;
        const int c0 = (tid & 15) * 4;
        const long e = (edgeBase + r < nEdges) ? edgeBase + r : nEdges - 1;
        *(uint2*)(SE + r * D_EDGE + c0) = cvt4_nt(ez + e * D_EDGE + c0);
    }
    __syncthreads();

    // ---- he tile: 16 edges x 16 cols per wave, K = 64 -> 2 WMMA ----
    const int n = wave * 16 + rl;                       // output column 0..127
    const unsigned short* wr = WeB + n * D_EDGE;
    const unsigned short* sr = SE + rl * D_EDGE;
    v8f acc = {};
#pragma unroll
    for (int kk = 0; kk < D_EDGE; kk += 32) {
        FragB16 a, b;
        a.u[0] = *(const uint4*)(sr + kk + 8 * half);
        a.u[1] = *(const uint4*)(sr + kk + 16 + 8 * half);
        b.u[0] = *(const uint4*)(wr + kk + 8 * half);
        b.u[1] = *(const uint4*)(wr + kk + 16 + 8 * half);
        acc = __builtin_amdgcn_wmma_f32_16x16x32_bf16(false, a.v, false, b.v,
                                                      (short)0, acc, false, false);
    }
    const float bias = be[n];
#pragma unroll
    for (int r = 0; r < 8; ++r) {
        long e = edgeBase + r + 8 * half;
        if (e < nEdges) {
            int d = dst[e];
            atomicAdd(&sum_e[(long)d * HDIM + n], acc[r] + bias);   // L2-resident
        }
    }

    // ---- gather hn[src] -> sum_n[dst]: 256 threads cover 2 edges x 128 cols ----
    const int col  = tid & 127;
    const int esub = tid >> 7;
    for (int e2 = 0; e2 < 16; e2 += 2) {
        long e = edgeBase + e2 + esub;
        if (e < nEdges) {
            int s = src[e], d = dst[e];
            atomicAdd(&sum_n[(long)d * HDIM + col], hn[(long)s * HDIM + col]);
        }
    }
    if (tid < 16) {
        long e = edgeBase + tid;
        if (e < nEdges) atomicAdd(&deg[dst[e]], 1.0f);
    }
}

// out = (concat(hn+sum_n, sum_e) @ Wr.T) / (deg+1) + br, zero where deg==0.
__global__ __launch_bounds__(256) void reduce_out(
    const float* __restrict__ hn, const float* __restrict__ sum_n,
    const float* __restrict__ sum_e, const float* __restrict__ deg,
    const unsigned short* __restrict__ WrB, const float* __restrict__ br,
    float* __restrict__ out, int nNodes) {
    __shared__ unsigned short S[16 * D_MSG];            // 16 x 256 bf16 = 8KB
    const int tid  = threadIdx.x;
    const int wave = tid >> 5;
    const int lane = tid & 31;
    const int rl   = lane & 15;
    const int half = lane >> 4;
    const int rowBase = blockIdx.x * 16;

#pragma unroll
    for (int r = 0; r < 16; ++r) {
        int node = (rowBase + r < nNodes) ? rowBase + r : nNodes - 1;
        float v;
        if (tid < HDIM) v = hn[(long)node * HDIM + tid] + sum_n[(long)node * HDIM + tid];
        else            v = sum_e[(long)node * HDIM + (tid - HDIM)];
        S[r * D_MSG + tid] = bf16s(v);
    }
    __syncthreads();

#pragma unroll
    for (int t = 0; t < 2; ++t) {
        const int ct = wave + t * 8;                    // col tile 0..15
        const int n  = ct * 16 + rl;                    // output column 0..255
        const unsigned short* wr = WrB + n * D_MSG;
        const unsigned short* sr = S + rl * D_MSG;
        v8f acc = {};
#pragma unroll
        for (int kk = 0; kk < D_MSG; kk += 32) {
            FragB16 a, b;
            a.u[0] = *(const uint4*)(sr + kk + 8 * half);
            a.u[1] = *(const uint4*)(sr + kk + 16 + 8 * half);
            b.u[0] = *(const uint4*)(wr + kk + 8 * half);
            b.u[1] = *(const uint4*)(wr + kk + 16 + 8 * half);
            acc = __builtin_amdgcn_wmma_f32_16x16x32_bf16(false, a.v, false, b.v,
                                                          (short)0, acc, false, false);
        }
        const float bias = br[n];
#pragma unroll
        for (int r = 0; r < 8; ++r) {
            int node = rowBase + r + 8 * half;
            if (node < nNodes) {
                float dg = deg[node];
                float val = (dg > 0.0f) ? (acc[r] / (dg + 1.0f) + bias) : 0.0f;
                __builtin_nontemporal_store(val, &out[(long)node * D_OUT + n]); // write-once
            }
        }
    }
}

extern "C" void kernel_launch(void* const* d_in, const int* in_sizes, int n_in,
                              void* d_out, int out_size, void* d_ws, size_t ws_size,
                              hipStream_t stream) {
    const float* x  = (const float*)d_in[0];
    const float* ez = (const float*)d_in[1];
    const int*   src = (const int*)d_in[2];
    const int*   dst = (const int*)d_in[3];
    const float* Wn = (const float*)d_in[4];
    const float* bn = (const float*)d_in[5];
    const float* We = (const float*)d_in[6];
    const float* be = (const float*)d_in[7];
    const float* Wr = (const float*)d_in[8];
    const float* br = (const float*)d_in[9];
    const int  nNodes = in_sizes[0] / D_NODE;
    const long nEdges = in_sizes[2];

    // carve workspace (sum_n, sum_e, deg contiguous at the front for one zero pass)
    char* ws = (char*)d_ws;
    size_t off = 0;
    auto carve = [&](size_t bytes) -> char* {
        char* p = ws + off;
        off = (off + bytes + 255) & ~(size_t)255;
        return p;
    };
    float* sum_n = (float*)carve((size_t)nNodes * HDIM * 4);
    float* sum_e = (float*)carve((size_t)nNodes * HDIM * 4);
    float* deg   = (float*)carve((size_t)nNodes * 4);
    const long zeroFloats = (long)(off / 4);            // sum_n..deg span
    float* hn = (float*)carve((size_t)nNodes * HDIM * 4);
    unsigned short* WnB = (unsigned short*)carve((size_t)HDIM * D_NODE * 2);
    unsigned short* WeB = (unsigned short*)carve((size_t)HDIM * D_EDGE * 2);
    unsigned short* WrB = (unsigned short*)carve((size_t)D_OUT * D_MSG * 2);
    (void)ws_size; (void)n_in; (void)out_size;

    zero_f32<<<2048, 256, 0, stream>>>(sum_n, zeroFloats);
    cvt_bf16<<<64, 256, 0, stream>>>(Wn, WnB, HDIM * D_NODE);
    cvt_bf16<<<16, 256, 0, stream>>>(We, WeB, HDIM * D_EDGE);
    cvt_bf16<<<128, 256, 0, stream>>>(Wr, WrB, D_OUT * D_MSG);

    hn_gemm<<<(nNodes + 15) / 16, 256, 0, stream>>>(x, WnB, bn, hn, nNodes);
    edge_msg<<<(int)((nEdges + 15) / 16), 256, 0, stream>>>(
        ez, WeB, be, src, dst, hn, sum_n, sum_e, deg, nEdges);
    reduce_out<<<(nNodes + 15) / 16, 256, 0, stream>>>(
        hn, sum_n, sum_e, deg, WrB, br, (float*)d_out, nNodes);
}
